// CrossAttentionLayer_21878563406365
// MI455X (gfx1250) — matrix operand order repaired and checked by reading, not compiled, not run on hardware
//
#include <hip/hip_runtime.h>
#include <hip/hip_bf16.h>
#include <math.h>

typedef __bf16 bf16;
typedef __bf16 v4bf  __attribute__((ext_vector_type(4)));
typedef __bf16 v8bf  __attribute__((ext_vector_type(8)));
typedef __bf16 v16bf __attribute__((ext_vector_type(16)));
typedef float  v4f   __attribute__((ext_vector_type(4)));
typedef float  v8f   __attribute__((ext_vector_type(8)));

#define C_DIM 512
#define H_NUM 8
#define D_DIM 64
#define L_SEQ 4096
#define B_NUM 4
#define M_TOT (B_NUM * L_SEQ)   /* 16384 rows for q/k/v projections */
#define NSEG  8                 /* split-K segments for KV stage    */

// ---------------------------------------------------------------------------
// WMMA fragment helpers (wave32, bf16 16x16x32, layouts per ISA 7.12.2)
//
// A (16x32, M x K): lane l holds row m = l&15; elems j=0..7 -> k = 8*(l>>4)+j,
//                   elems j=8..15 -> k = 16 + 8*(l>>4) + (j-8).
// B (32x16, K x N): lane l holds col n = l&15; same k order as A.
// C/D (16x16 f32) : lane l col n = l&15; VGPR i row m = i + 8*(l>>4).
// ---------------------------------------------------------------------------

__device__ __forceinline__ v16bf load_frag_a(const bf16* row, int koff) {
  v8bf lo = *reinterpret_cast<const v8bf*>(row + koff);
  v8bf hi = *reinterpret_cast<const v8bf*>(row + 16 + koff);
  return __builtin_shufflevector(lo, hi, 0,1,2,3,4,5,6,7,8,9,10,11,12,13,14,15);
}

__device__ __forceinline__ v16bf load_frag_b(const bf16* row, int kbase) {
  v8bf lo = *reinterpret_cast<const v8bf*>(row + kbase / 2);
  v8bf hi = *reinterpret_cast<const v8bf*>(row + 16 + kbase / 2);
  return __builtin_shufflevector(lo, hi, 0,1,2,3,4,5,6,7,8,9,10,11,12,13,14,15);
}

__device__ __forceinline__ v8f wmma_bf16(v16bf a, v16bf b, v8f c) {
  return __builtin_amdgcn_wmma_f32_16x16x32_bf16(false, a, false, b, (short)0, c,
                                                 false, false);
}

// ---------------------------------------------------------------------------
// Stage 1: projection GEMM  out[m,n] = act( sum_k In[m,k]*W[n,k] + bias[n] )
//   In: [M_TOT, 512] f32; W: [512(out), 512(in)] f32, both row-major.
//   Block tile 128x128, 8 waves in 4(M)x2(N); each wave 32(M)x64(N):
//   2 A-frags x 4 B-frags -> 8 WMMAs per 32-wide K chunk.
//   All tile staging uses 128-bit global loads (float4) + vector bf16 stores.
// ---------------------------------------------------------------------------
__global__ __launch_bounds__(256) void proj_kernel(
    const float* __restrict__ A, const float* __restrict__ W,
    const float* __restrict__ bias, bf16* __restrict__ Out, int feat)
{
  __shared__ bf16 As[128][40];   // [m][k], stride 40 elems = 80B (16B aligned)
  __shared__ bf16 Bs[128][40];   // [n][k]

  const int t    = threadIdx.x;
  const int lane = t & 31;
  const int wid  = t >> 5;
  const int m0   = blockIdx.x * 128;
  const int n0   = blockIdx.y * 128;
  const int wm   = wid >> 1;     // 0..3 -> 32-row strip
  const int wn   = wid & 1;      // 0..1 -> 64-col strip
  const int mrow = lane & 15;
  const int koff = (lane >> 4) * 8;
  const int kb16 = (lane >> 4) * 16;

  // tile loader mapping: 1024 float4 per tile, 4 per thread
  const int lr  = t >> 3;        // rows handled: lr, lr+32, lr+64, lr+96
  const int lc4 = (t & 7) * 4;   // k column group (float4)

  v8f acc[2][4] = {};

  for (int kt = 0; kt < C_DIM / 32; ++kt) {
    const int k0 = kt * 32;
    #pragma unroll
    for (int i = 0; i < 4; ++i) {
      const int r = lr + i * 32;
      v4f av = *reinterpret_cast<const v4f*>(&A[(size_t)(m0 + r) * C_DIM + k0 + lc4]);
      v4f wv = *reinterpret_cast<const v4f*>(&W[(size_t)(n0 + r) * C_DIM + k0 + lc4]);
      *reinterpret_cast<v4bf*>(&As[r][lc4]) = __builtin_convertvector(av, v4bf);
      *reinterpret_cast<v4bf*>(&Bs[r][lc4]) = __builtin_convertvector(wv, v4bf);
    }
    __syncthreads();

    if (kt + 1 < C_DIM / 32)  // hint next A tile toward the WGP caches
      __builtin_prefetch(&A[(size_t)(m0 + lr) * C_DIM + k0 + 32 + lc4], 0, 3);

    v16bf afrag[2], bfrag[4];
    #pragma unroll
    for (int mt = 0; mt < 2; ++mt)
      afrag[mt] = load_frag_a(&As[wm * 32 + mt * 16 + mrow][0], koff);
    #pragma unroll
    for (int nt = 0; nt < 4; ++nt)
      bfrag[nt] = load_frag_b(&Bs[wn * 64 + nt * 16 + mrow][0], kb16);
    #pragma unroll
    for (int mt = 0; mt < 2; ++mt)
      #pragma unroll
      for (int nt = 0; nt < 4; ++nt)
        acc[mt][nt] = wmma_bf16(afrag[mt], bfrag[nt], acc[mt][nt]);
    __syncthreads();
  }

  #pragma unroll
  for (int mt = 0; mt < 2; ++mt)
    #pragma unroll
    for (int nt = 0; nt < 4; ++nt) {
      const int n = n0 + wn * 64 + nt * 16 + (lane & 15);
      const float bn = bias[n];
      #pragma unroll
      for (int i = 0; i < 8; ++i) {
        const int m = m0 + wm * 32 + mt * 16 + i + 8 * (lane >> 4);
        float v = acc[mt][nt][i] + bn;
        if (feat) v = (v > 0.f) ? (v + 1.f) : __expf(v);
        Out[(size_t)m * C_DIM + n] = (bf16)v;
      }
    }
}

// ---------------------------------------------------------------------------
// Stage 2: per (b,h,seg) partial  KV[d,e] += sum_s K[s,d]*V[s,e],
//          Ksum[d] += sum_s K[s,d].  64x64 output, K-range = 512 per segment.
//   LDS tiles stored transposed ([d][s], [e][s]) so WMMA fragment gathers are
//   contiguous 16B LDS reads. Tile staging: one v8bf (128-bit) global load per
//   thread per tile. 8 waves: wave w -> dtile=w>>1, e-half=w&1 (2 tiles).
// ---------------------------------------------------------------------------
__global__ __launch_bounds__(256) void kv_kernel(
    const bf16* __restrict__ Kf, const bf16* __restrict__ Vf,
    float* __restrict__ KVp, float* __restrict__ Ksp)
{
  __shared__ bf16 Kt[64][40];    // [d][s_local]
  __shared__ bf16 Vt[64][40];    // [e][s_local]
  __shared__ float kred[4][64];

  const int t = threadIdx.x, lane = t & 31, wid = t >> 5;
  const int bh = blockIdx.x, seg = blockIdx.y;
  const int b = bh >> 3, h = bh & 7;
  const int dt = wid >> 1, eb = wid & 1;
  const int mrow = lane & 15;
  const int koff = (lane >> 4) * 8, kb16 = (lane >> 4) * 16;
  const int myd = t & 63, sg = t >> 6;
  const int lsl = t >> 3;          // s_local 0..31
  const int ld8 = (t & 7) * 8;     // d group of 8
  const size_t base = (size_t)b * L_SEQ * C_DIM + (size_t)h * D_DIM;

  v8f acc[2] = {};
  float ks = 0.f;

  for (int sc = 0; sc < 16; ++sc) {
    const int s0 = seg * 512 + sc * 32;
    {
      const size_t g = base + (size_t)(s0 + lsl) * C_DIM + ld8;
      v8bf kv8 = *reinterpret_cast<const v8bf*>(&Kf[g]);
      v8bf vv8 = *reinterpret_cast<const v8bf*>(&Vf[g]);
      #pragma unroll
      for (int j = 0; j < 8; ++j) {
        Kt[ld8 + j][lsl] = kv8[j];
        Vt[ld8 + j][lsl] = vv8[j];
      }
    }
    __syncthreads();

    #pragma unroll
    for (int j = 0; j < 8; ++j) ks += (float)Kt[myd][sg * 8 + j];

    v16bf af = load_frag_a(&Kt[dt * 16 + mrow][0], koff);
    #pragma unroll
    for (int nt = 0; nt < 2; ++nt) {
      v16bf bfg = load_frag_b(&Vt[(eb * 2 + nt) * 16 + mrow][0], kb16);
      acc[nt] = wmma_bf16(af, bfg, acc[nt]);
    }
    __syncthreads();
  }

  kred[sg][myd] = ks;
  __syncthreads();
  if (t < 64)
    Ksp[((size_t)bh * NSEG + seg) * 64 + t] =
        kred[0][t] + kred[1][t] + kred[2][t] + kred[3][t];

  float* outp = KVp + ((size_t)bh * NSEG + seg) * 4096;
  #pragma unroll
  for (int nt = 0; nt < 2; ++nt) {
    const int e = (eb * 2 + nt) * 16 + (lane & 15);
    #pragma unroll
    for (int i = 0; i < 8; ++i) {
      const int d = dt * 16 + i + 8 * (lane >> 4);
      outp[d * 64 + e] = acc[nt][i];
    }
  }
}

// ---------------------------------------------------------------------------
// Stage 2b: reduce NSEG partials -> KV stored TRANSPOSED [e][d] in bf16 (so
// stage 3 can stage it with pure 128-bit loads/stores) + Ksum (f32).
// ---------------------------------------------------------------------------
__global__ __launch_bounds__(256) void reduce_kernel(
    const float* __restrict__ KVp, const float* __restrict__ Ksp,
    bf16* __restrict__ KVb, float* __restrict__ Ksum)
{
  const int bh = blockIdx.x, t = threadIdx.x;
  #pragma unroll
  for (int j = 0; j < 16; ++j) {
    int idx = t + j * 256;          // idx = d*64 + e in the partials
    int d = idx >> 6, e = idx & 63;
    float s = 0.f;
    #pragma unroll
    for (int seg = 0; seg < NSEG; ++seg)
      s += KVp[((size_t)bh * NSEG + seg) * 4096 + idx];
    KVb[(size_t)bh * 4096 + e * 64 + d] = (bf16)s;   // transposed store
  }
  if (t < 64) {
    float s = 0.f;
    #pragma unroll
    for (int seg = 0; seg < NSEG; ++seg)
      s += Ksp[((size_t)bh * NSEG + seg) * 64 + t];
    Ksum[bh * 64 + t] = s;
  }
}

// ---------------------------------------------------------------------------
// Stage 3: out[l,e] = (Q[l,:] @ KV[:, e]) / (Q[l,:] . Ksum + eps)
//   (the reference's v/S scaling and final *S cancel exactly)
//   Block: 128 L-rows x full head (64). 8 waves, wave w -> rows w*16..+16,
//   4 e-tiles x 2 k-chunks of WMMA. All LDS staging via 128-bit accesses.
// ---------------------------------------------------------------------------
__global__ __launch_bounds__(256) void out_kernel(
    const bf16* __restrict__ Qf, const bf16* __restrict__ KVb,
    const float* __restrict__ Ksum, float* __restrict__ Out)
{
  __shared__ bf16 Qs[128][72];    // [m][d], stride 72 elems = 144B (16B aligned)
  __shared__ bf16 KVt[64][72];    // [e][d]
  __shared__ float ksums[64];
  __shared__ float zpart[128][2];
  __shared__ float zs[128];

  const int t = threadIdx.x, lane = t & 31, wid = t >> 5;
  const int lx = blockIdx.x, bh = blockIdx.y;
  const int b = bh >> 3, h = bh & 7;
  const int l0 = lx * 128;
  const size_t qbase = ((size_t)b * L_SEQ + l0) * C_DIM + (size_t)h * D_DIM;

  // KV tile: already [e][d] in global -> straight 128-bit copy
  #pragma unroll
  for (int i = 0; i < 2; ++i) {
    int idx = t + i * 256;                 // 512 v8bf total
    int e = idx >> 3, d8 = (idx & 7) * 8;
    *reinterpret_cast<v8bf*>(&KVt[e][d8]) =
        *reinterpret_cast<const v8bf*>(&KVb[(size_t)bh * 4096 + e * 64 + d8]);
  }
  if (t < 64) ksums[t] = Ksum[bh * 64 + t];
  // Q tile 128x64: 1024 v8bf, 4 per thread, contiguous both sides
  #pragma unroll
  for (int i = 0; i < 4; ++i) {
    int idx = t + i * 256;
    int r = idx >> 3, d8 = (idx & 7) * 8;
    *reinterpret_cast<v8bf*>(&Qs[r][d8]) =
        *reinterpret_cast<const v8bf*>(&Qf[qbase + (size_t)r * C_DIM + d8]);
  }
  __syncthreads();

  {  // per-row normalizer
    const int r = t >> 1, half = t & 1;
    float p = 0.f;
    #pragma unroll
    for (int j = 0; j < 32; ++j)
      p += (float)Qs[r][half * 32 + j] * ksums[half * 32 + j];
    zpart[r][half] = p;
  }
  __syncthreads();
  if (t < 128) zs[t] = 1.f / (zpart[t][0] + zpart[t][1] + 1e-6f);
  __syncthreads();

  const int mrow = lane & 15;
  const int koff = (lane >> 4) * 8, kb16 = (lane >> 4) * 16;
  const int mtile = wid * 16;

  v8f acc[4] = {};
  v16bf af[2];
  #pragma unroll
  for (int kc = 0; kc < 2; ++kc)
    af[kc] = load_frag_a(&Qs[mtile + mrow][kc * 32], koff);
  #pragma unroll
  for (int et = 0; et < 4; ++et)
    #pragma unroll
    for (int kc = 0; kc < 2; ++kc) {
      v16bf bfg = load_frag_b(&KVt[et * 16 + mrow][kc * 32], kb16);
      acc[et] = wmma_bf16(af[kc], bfg, acc[et]);
    }

  #pragma unroll
  for (int et = 0; et < 4; ++et) {
    const int e = et * 16 + (lane & 15);
    #pragma unroll
    for (int i = 0; i < 8; ++i) {
      const int m = mtile + i + 8 * (lane >> 4);
      Out[((size_t)b * L_SEQ + l0 + m) * C_DIM + (size_t)h * D_DIM + e] =
          acc[et][i] * zs[m];
    }
  }
}

// ---------------------------------------------------------------------------
// Launch: inputs = x, guidance, Wq, bq, Wk, bk, Wv, bv (all f32). Out f32.
// Workspace: qf/kf/vf bf16 (16 MB each), KV partials, reduced KV/Ksum. ~53 MB.
// ---------------------------------------------------------------------------
extern "C" void kernel_launch(void* const* d_in, const int* in_sizes, int n_in,
                              void* d_out, int out_size, void* d_ws, size_t ws_size,
                              hipStream_t stream)
{
  const float* x  = (const float*)d_in[0];
  const float* g  = (const float*)d_in[1];
  const float* Wq = (const float*)d_in[2];
  const float* bq = (const float*)d_in[3];
  const float* Wk = (const float*)d_in[4];
  const float* bk = (const float*)d_in[5];
  const float* Wv = (const float*)d_in[6];
  const float* bv = (const float*)d_in[7];
  float* out = (float*)d_out;

  char* ws = (char*)d_ws;
  const size_t NQ = (size_t)M_TOT * C_DIM;              // 8,388,608 elems
  const size_t OFF_QF  = 0;
  const size_t OFF_KF  = OFF_QF + NQ * sizeof(bf16);
  const size_t OFF_VF  = OFF_KF + NQ * sizeof(bf16);
  const size_t OFF_KVP = OFF_VF + NQ * sizeof(bf16);
  const size_t OFF_KSP = OFF_KVP + (size_t)32 * NSEG * 4096 * sizeof(float);
  const size_t OFF_KVB = OFF_KSP + (size_t)32 * NSEG * 64 * sizeof(float);
  const size_t OFF_KSM = OFF_KVB + (size_t)32 * 4096 * sizeof(bf16);

  bf16*  qf   = (bf16*)(ws + OFF_QF);
  bf16*  kf   = (bf16*)(ws + OFF_KF);
  bf16*  vf   = (bf16*)(ws + OFF_VF);
  float* KVp  = (float*)(ws + OFF_KVP);
  float* Ksp  = (float*)(ws + OFF_KSP);
  bf16*  KVb  = (bf16*)(ws + OFF_KVB);
  float* Ksum = (float*)(ws + OFF_KSM);

  dim3 gp(M_TOT / 128, C_DIM / 128);
  proj_kernel<<<gp, 256, 0, stream>>>(g, Wq, bq, qf, 1);
  proj_kernel<<<gp, 256, 0, stream>>>(x, Wk, bk, kf, 1);
  proj_kernel<<<gp, 256, 0, stream>>>(x, Wv, bv, vf, 0);
  kv_kernel<<<dim3(B_NUM * H_NUM, NSEG), 256, 0, stream>>>(kf, vf, KVp, Ksp);
  reduce_kernel<<<B_NUM * H_NUM, 256, 0, stream>>>(KVp, Ksp, KVb, Ksum);
  out_kernel<<<dim3(L_SEQ / 128, B_NUM * H_NUM), 256, 0, stream>>>(qf, KVb, Ksum, out);
}
